// CogvlmVisionExpertAttention_67138928771201
// MI455X (gfx1250) — matrix operand order repaired and checked by reading, not compiled, not run on hardware
//
#include <hip/hip_runtime.h>
#include <hip/hip_bf16.h>
#include <math.h>

// Problem constants (B=1)
#define S_LEN 2048
#define HID   4096
#define NHEAD 32
#define HDIM  128
#define QKV_N 12288   // 3*HID

typedef __bf16 bf16_t;
typedef __attribute__((ext_vector_type(8)))  __bf16 v8bf;
typedef __attribute__((ext_vector_type(16))) __bf16 v16bf;
typedef __attribute__((ext_vector_type(8)))  float  v8f;

__device__ __forceinline__ v16bf cat8(v8bf lo, v8bf hi) {
  return __builtin_shufflevector(lo, hi, 0,1,2,3,4,5,6,7,8,9,10,11,12,13,14,15);
}

// CDNA5 async copy: 16B global -> LDS, tracked by ASYNCcnt.
__device__ __forceinline__ void async_b128(unsigned lds_off, const void* gaddr) {
  asm volatile("global_load_async_to_lds_b128 %0, %1, off"
               :: "v"(lds_off), "v"(gaddr) : "memory");
}
__device__ __forceinline__ void wait_async0() {
  asm volatile("s_wait_asynccnt 0" ::: "memory");
}
// Async loads complete in order: <=4 outstanding => the previously issued
// 4 copies (current tile) have landed while the next 4 keep streaming.
__device__ __forceinline__ void wait_async4() {
  asm volatile("s_wait_asynccnt 4" ::: "memory");
}
__device__ __forceinline__ void wait_ds0() {
  asm volatile("s_wait_dscnt 0" ::: "memory");
}

__device__ __forceinline__ v8f wmma_bf16(v16bf a, v16bf b, v8f c) {
  return __builtin_amdgcn_wmma_f32_16x16x32_bf16(false, a, false, b, (short)0, c, false, false);
}

// ---------------------------------------------------------------------------
// 1) Routing: vision token i iff tt[i]==1 && tt[i+1]==1 (last is language).
// ---------------------------------------------------------------------------
__global__ void build_routing(const int* __restrict__ tt,
                              int* __restrict__ vis, int* __restrict__ lng,
                              int* __restrict__ counts) {
  int nv = 0, nl = 0;
  for (int i = 0; i < S_LEN; ++i) {
    int f = (i < S_LEN - 1) && (tt[i] == 1) && (tt[i + 1] == 1);
    if (f) vis[nv++] = i; else lng[nl++] = i;
  }
  counts[0] = nv; counts[1] = nl;
}

// ---------------------------------------------------------------------------
// 2) fp32 -> bf16 elementwise
// ---------------------------------------------------------------------------
__global__ void cvt_bf16(const float* __restrict__ x, bf16_t* __restrict__ y, int n) {
  int i = blockIdx.x * blockDim.x + threadIdx.x;
  if (i < n) y[i] = (bf16_t)x[i];
}

// ---------------------------------------------------------------------------
// 3) W[K][N] fp32 -> Wt[N][K] bf16  (LDS-tiled transpose, 32x32 tiles)
// ---------------------------------------------------------------------------
__global__ __launch_bounds__(256) void transpose_cvt(
    const float* __restrict__ W, bf16_t* __restrict__ Wt, int K, int N) {
  __shared__ float t[32][33];
  const int n0 = blockIdx.x * 32, k0 = blockIdx.y * 32;
  const int tx = threadIdx.x & 31, ty = threadIdx.x >> 5;
#pragma unroll
  for (int i = 0; i < 32; i += 8)
    t[ty + i][tx] = W[(size_t)(k0 + ty + i) * N + n0 + tx];
  __syncthreads();
#pragma unroll
  for (int i = 0; i < 32; i += 8)
    Wt[(size_t)(n0 + ty + i) * K + k0 + tx] = (bf16_t)t[tx][ty + i];
}

// ---------------------------------------------------------------------------
// 4) Routed GEMM: Out[idx[m], n] = sum_k A[idx[m], k] * Wt[n, k]
//    128x128 tile / workgroup, 8 waves as 4(m) x 2(n), each 32x64.
//    Double-buffered async-to-LDS pipeline on ASYNCcnt.
// ---------------------------------------------------------------------------
__global__ __launch_bounds__(256) void gemm_routed(
    const bf16_t* __restrict__ A, const bf16_t* __restrict__ Bt,
    float* __restrict__ Out, const int* __restrict__ idx,
    const int* __restrict__ countp, int K, int lda, int ldo) {
  const int count = countp[0];
  const int m0 = blockIdx.y * 128;
  if (m0 >= count) return;
  const int n0 = blockIdx.x * 128;

  __shared__ alignas(16) bf16_t sA[2][128 * 40];  // 32 data + 8 pad bf16 / row
  __shared__ alignas(16) bf16_t sB[2][128 * 40];

  const int tid  = threadIdx.x;
  const int lane = tid & 31;
  const int wv   = tid >> 5;
  const int wm   = (wv & 3) * 32;
  const int wn   = (wv >> 2) * 64;
  const int half = lane >> 4;
  const int l16  = lane & 15;

  const unsigned sA0[2] = { (unsigned)(size_t)&sA[0][0], (unsigned)(size_t)&sA[1][0] };
  const unsigned sB0[2] = { (unsigned)(size_t)&sB[0][0], (unsigned)(size_t)&sB[1][0] };

  // Per-thread fill tasks: 2 A-chunks + 2 B-chunks (16B each) per K-step.
  const bf16_t* agp[2]; unsigned aoff[2];
  const bf16_t* bgp[2]; unsigned boff[2];
#pragma unroll
  for (int t = 0; t < 2; ++t) {
    const int id = tid + t * 256;
    const int row = id >> 2, ch = id & 3;
    const int slot = m0 + row;
    const int rg = (slot < count) ? idx[slot] : 0;   // clamp: masked on store
    agp[t]  = A + (size_t)rg * lda + ch * 8;
    aoff[t] = row * 80 + ch * 16;
    bgp[t]  = Bt + (size_t)(n0 + row) * K + ch * 8;
    boff[t] = row * 80 + ch * 16;
  }

  v8f acc[2][4];
#pragma unroll
  for (int i = 0; i < 2; ++i)
#pragma unroll
    for (int j = 0; j < 4; ++j)
#pragma unroll
      for (int e = 0; e < 8; ++e) acc[i][j][e] = 0.f;

  // ---- prologue: stage K-step 0 into buffer 0
#pragma unroll
  for (int t = 0; t < 2; ++t) {
    async_b128(sA0[0] + aoff[t], agp[t]);
    async_b128(sB0[0] + boff[t], bgp[t]);
  }

  const int nsteps = K / 32;
  for (int s = 0; s < nsteps; ++s) {
    const int cur = s & 1;
    const int k0  = s * 32;

    if (s + 1 < nsteps) {
      // buffer cur^1 was fully consumed at step s-1 (trailing barrier below)
      const int nk = k0 + 32;
#pragma unroll
      for (int t = 0; t < 2; ++t) {
        async_b128(sA0[cur ^ 1] + aoff[t], agp[t] + nk);
        async_b128(sB0[cur ^ 1] + boff[t], bgp[t] + nk);
      }
      wait_async4();   // current tile's 4 copies landed; next 4 keep streaming
    } else {
      wait_async0();
    }
    __syncthreads();

    v16bf af[2], bfm[4];
#pragma unroll
    for (int mt = 0; mt < 2; ++mt) {           // A 16x32: elems0-7 K=8h.., 8-15 K=16+8h..
      const bf16_t* p = &sA[cur][(wm + mt * 16 + l16) * 40 + 8 * half];
      af[mt] = cat8(*(const v8bf*)p, *(const v8bf*)(p + 16));
    }
#pragma unroll
    for (int nt = 0; nt < 4; ++nt) {           // B 32x16: elems0-7 K=8h.., 8-15 K=16+8h..
      const bf16_t* p = &sB[cur][(wn + nt * 16 + l16) * 40 + 8 * half];
      bfm[nt] = cat8(*(const v8bf*)p, *(const v8bf*)(p + 16));
    }
#pragma unroll
    for (int mt = 0; mt < 2; ++mt)
#pragma unroll
      for (int nt = 0; nt < 4; ++nt)
        acc[mt][nt] = wmma_bf16(af[mt], bfm[nt], acc[mt][nt]);

    __syncthreads();   // all waves done with buffer cur before it is refilled
  }

#pragma unroll
  for (int mt = 0; mt < 2; ++mt)
#pragma unroll
    for (int r = 0; r < 8; ++r) {
      const int slot = m0 + wm + mt * 16 + r + 8 * half;   // C: M = r + 8*half
      if (slot < count) {
        const size_t orow = (size_t)idx[slot] * ldo;
#pragma unroll
        for (int nt = 0; nt < 4; ++nt)
          Out[orow + n0 + wn + nt * 16 + l16] = acc[mt][nt][r];
      }
    }
}

// ---------------------------------------------------------------------------
// 5) RoPE + head split.  q scaled by 1/sqrt(HD) (folded qk scale).
//    Q,K -> [NH][S][HD] bf16 ; V -> transposed [NH][HD][S] bf16 (for PV frags)
// ---------------------------------------------------------------------------
__global__ __launch_bounds__(128) void rope_split(
    const float* __restrict__ qkv, const int* __restrict__ pos_ids,
    bf16_t* __restrict__ Qb, bf16_t* __restrict__ Kb, bf16_t* __restrict__ Vt) {
  const int s = blockIdx.x, h = blockIdx.y, d = threadIdx.x;
  const float* row = qkv + (size_t)s * QKV_N + h * HDIM;
  const float p = (float)pos_ids[s];
  const int fi = d & 63;
  const float invf = __expf(-(float)fi * 0.14391156831f);  // 10000^(-2*fi/128)
  const float ang = p * invf;
  const float c = __cosf(ang), sn = __sinf(ang);
  const int dp = (d < 64) ? d + 64 : d - 64;
  const float sgn = (d < 64) ? -1.f : 1.f;
  const float qv = row[d],        qq = row[dp];
  const float kv = row[HID + d],  kk = row[HID + dp];
  const float vv = row[2 * HID + d];
  const float qo = (qv * c + sgn * qq * sn) * 0.08838834765f;  // 1/sqrt(128)
  const float ko = (kv * c + sgn * kk * sn);
  Qb[((size_t)h * S_LEN + s) * HDIM + d] = (bf16_t)qo;
  Kb[((size_t)h * S_LEN + s) * HDIM + d] = (bf16_t)ko;
  Vt[(size_t)h * HDIM * S_LEN + (size_t)d * S_LEN + s] = (bf16_t)vv;
}

// ---------------------------------------------------------------------------
// 6) Flash attention, causal.  Block = 4 waves; each wave owns 16 q-rows of a
//    64-row q-block.  K/V tiles staged with async-to-LDS; QK^T and PV on WMMA;
//    online softmax with 16-lane shuffle reductions (wave32).
// ---------------------------------------------------------------------------
__global__ __launch_bounds__(128) void attn_fwd(
    const bf16_t* __restrict__ Qb, const bf16_t* __restrict__ Kb,
    const bf16_t* __restrict__ Vt, bf16_t* __restrict__ Ctx) {
  const int qb = blockIdx.x, h = blockIdx.y;
  __shared__ alignas(16) bf16_t sK[64 * 144];   // [key][d], 128 data + 16 pad
  __shared__ alignas(16) bf16_t sV[128 * 72];   // [d][key], 64 data + 8 pad
  __shared__ alignas(16) bf16_t sP[4 * 16 * 72];// per-wave P^T staging

  const int tid = threadIdx.x, lane = tid & 31, wv = tid >> 5;
  const int half = lane >> 4, l16 = lane & 15;
  const int q0w = qb * 64 + wv * 16;

  const bf16_t* Kh = Kb + (size_t)h * S_LEN * HDIM;
  const bf16_t* Vh = Vt + (size_t)h * HDIM * S_LEN;
  const unsigned sK0 = (unsigned)(size_t)&sK[0];
  const unsigned sV0 = (unsigned)(size_t)&sV[0];

  // Q fragments held in registers for the whole kernel (4 k-steps of 32).
  v16bf qf[4];
  {
    const bf16_t* qrow = Qb + ((size_t)h * S_LEN + q0w + l16) * HDIM;
#pragma unroll
    for (int kk = 0; kk < 4; ++kk) {
      const bf16_t* p = qrow + kk * 32 + 8 * half;
      qf[kk] = cat8(*(const v8bf*)p, *(const v8bf*)(p + 16));
    }
  }

  float mi[8], li[8];
  v8f o[8];
#pragma unroll
  for (int r = 0; r < 8; ++r) { mi[r] = -3.0e38f; li[r] = 0.f; }
#pragma unroll
  for (int dt = 0; dt < 8; ++dt)
#pragma unroll
    for (int e = 0; e < 8; ++e) o[dt][e] = 0.f;

  for (int kb = 0; kb <= qb; ++kb) {   // causal: only blocks with keys <= qmax
    const int kbase = kb * 64;
    __syncthreads();
#pragma unroll
    for (int i = 0; i < 8; ++i) {      // stage K tile 64x128
      const int id = tid + i * 128, r = id >> 4, c = id & 15;
      async_b128(sK0 + r * 288 + c * 16, Kh + (size_t)(kbase + r) * HDIM + c * 8);
    }
#pragma unroll
    for (int i = 0; i < 8; ++i) {      // stage V^T tile 128x64
      const int id = tid + i * 128, dd = id >> 3, c = id & 7;
      async_b128(sV0 + dd * 144 + c * 16, Vh + (size_t)dd * S_LEN + kbase + c * 8);
    }
    wait_async0();
    __syncthreads();

    // scores S = Q K^T  (16q x 64k per wave)
    v8f sc[4];
#pragma unroll
    for (int nt = 0; nt < 4; ++nt)
#pragma unroll
      for (int e = 0; e < 8; ++e) sc[nt][e] = 0.f;
#pragma unroll
    for (int nt = 0; nt < 4; ++nt)
#pragma unroll
      for (int kk = 0; kk < 4; ++kk) {
        const bf16_t* p = &sK[(nt * 16 + l16) * 144 + kk * 32 + 8 * half];
        v16bf bb = cat8(*(const v8bf*)p, *(const v8bf*)(p + 16));
        sc[nt] = wmma_bf16(qf[kk], bb, sc[nt]);
      }

    // causal mask + online softmax (per q-row; row = 16 lanes of one group)
    bf16_t* pw = &sP[wv * 16 * 72];
#pragma unroll
    for (int r = 0; r < 8; ++r) {
      const int q = q0w + r + 8 * half;
      float mx = -3.0e38f;
#pragma unroll
      for (int nt = 0; nt < 4; ++nt) {
        const int key = kbase + nt * 16 + l16;
        float v = sc[nt][r] + ((key > q) ? -1e9f : 0.f);
        sc[nt][r] = v;
        mx = fmaxf(mx, v);
      }
#pragma unroll
      for (int dlt = 1; dlt < 16; dlt <<= 1)
        mx = fmaxf(mx, __shfl_xor(mx, dlt, 32));
      const float mnew  = fmaxf(mi[r], mx);
      const float alpha = __expf(mi[r] - mnew);
      float ps = 0.f;
#pragma unroll
      for (int nt = 0; nt < 4; ++nt) {
        const float pex = __expf(sc[nt][r] - mnew);
        ps += pex;
        pw[(r + 8 * half) * 72 + nt * 16 + l16] = (bf16_t)pex;  // P -> LDS (transpose)
      }
#pragma unroll
      for (int dlt = 1; dlt < 16; dlt <<= 1)
        ps += __shfl_xor(ps, dlt, 32);
      li[r] = li[r] * alpha + ps;
      mi[r] = mnew;
#pragma unroll
      for (int dt = 0; dt < 8; ++dt) o[dt][r] *= alpha;
    }
    wait_ds0();   // P stores visible to this wave's A-fragment loads

    // O += P (16x64) x V (64x128)
#pragma unroll
    for (int kk2 = 0; kk2 < 2; ++kk2) {
      const bf16_t* pr = pw + l16 * 72 + kk2 * 32 + 8 * half;
      v16bf pa = cat8(*(const v8bf*)pr, *(const v8bf*)(pr + 16));
#pragma unroll
      for (int dt = 0; dt < 8; ++dt) {
        const bf16_t* vp = &sV[(dt * 16 + l16) * 72 + kk2 * 32 + 8 * half];
        v16bf vb = cat8(*(const v8bf*)vp, *(const v8bf*)(vp + 16));
        o[dt] = wmma_bf16(pa, vb, o[dt]);
      }
    }
  }

  // normalize and store ctx as bf16 [S][H] (col = h*HD + d)
#pragma unroll
  for (int r = 0; r < 8; ++r) {
    const int q = q0w + r + 8 * half;
    const float inv = 1.0f / li[r];
#pragma unroll
    for (int dt = 0; dt < 8; ++dt)
      Ctx[(size_t)q * HID + h * HDIM + dt * 16 + l16] = (bf16_t)(o[dt][r] * inv);
  }
}

// ---------------------------------------------------------------------------
// Launcher
// ---------------------------------------------------------------------------
extern "C" void kernel_launch(void* const* d_in, const int* in_sizes, int n_in,
                              void* d_out, int out_size, void* d_ws, size_t ws_size,
                              hipStream_t stream) {
  (void)in_sizes; (void)n_in; (void)out_size; (void)ws_size;
  const float* x   = (const float*)d_in[0];
  const int*   tt  = (const int*)d_in[1];   // token_type_ids
  const int*   pos = (const int*)d_in[2];   // position_ids
  // d_in[3] = attention_mask: analytically causal (-1e9 upper tri) -> re-derived
  const float* wvq = (const float*)d_in[4];
  const float* wlq = (const float*)d_in[5];
  const float* wvd = (const float*)d_in[6];
  const float* wld = (const float*)d_in[7];
  float* out = (float*)d_out;

  char* w = (char*)d_ws; size_t off = 0;
  auto alloc = [&](size_t b) { char* p = w + off; off = (off + b + 255) & ~(size_t)255; return p; };
  int*    vis_idx = (int*)alloc((size_t)S_LEN * 4);
  int*    lng_idx = (int*)alloc((size_t)S_LEN * 4);
  int*    counts  = (int*)alloc(8);
  bf16_t* x_bf    = (bf16_t*)alloc((size_t)S_LEN * HID * 2);
  bf16_t* wtq_v   = (bf16_t*)alloc((size_t)QKV_N * HID * 2);
  bf16_t* wtq_l   = (bf16_t*)alloc((size_t)QKV_N * HID * 2);
  bf16_t* wtd_v   = (bf16_t*)alloc((size_t)HID * HID * 2);
  bf16_t* wtd_l   = (bf16_t*)alloc((size_t)HID * HID * 2);
  float*  qkv     = (float*) alloc((size_t)S_LEN * QKV_N * 4);
  bf16_t* q_bf    = (bf16_t*)alloc((size_t)NHEAD * S_LEN * HDIM * 2);
  bf16_t* k_bf    = (bf16_t*)alloc((size_t)NHEAD * S_LEN * HDIM * 2);
  bf16_t* v_tr    = (bf16_t*)alloc((size_t)NHEAD * HDIM * S_LEN * 2);
  bf16_t* ctx     = (bf16_t*)alloc((size_t)S_LEN * HID * 2);

  build_routing<<<1, 1, 0, stream>>>(tt, vis_idx, lng_idx, counts);
  cvt_bf16<<<(S_LEN * HID) / 256, 256, 0, stream>>>(x, x_bf, S_LEN * HID);

  transpose_cvt<<<dim3(QKV_N / 32, HID / 32), 256, 0, stream>>>(wvq, wtq_v, HID, QKV_N);
  transpose_cvt<<<dim3(QKV_N / 32, HID / 32), 256, 0, stream>>>(wlq, wtq_l, HID, QKV_N);
  transpose_cvt<<<dim3(HID / 32,   HID / 32), 256, 0, stream>>>(wvd, wtd_v, HID, HID);
  transpose_cvt<<<dim3(HID / 32,   HID / 32), 256, 0, stream>>>(wld, wtd_l, HID, HID);

  dim3 gq(QKV_N / 128, S_LEN / 128);
  gemm_routed<<<gq, 256, 0, stream>>>(x_bf, wtq_v, qkv, vis_idx, counts + 0, HID, HID, QKV_N);
  gemm_routed<<<gq, 256, 0, stream>>>(x_bf, wtq_l, qkv, lng_idx, counts + 1, HID, HID, QKV_N);

  rope_split<<<dim3(S_LEN, NHEAD), 128, 0, stream>>>(qkv, pos, q_bf, k_bf, v_tr);
  attn_fwd<<<dim3(S_LEN / 64, NHEAD), 128, 0, stream>>>(q_bf, k_bf, v_tr, ctx);

  dim3 gd(HID / 128, S_LEN / 128);
  gemm_routed<<<gd, 256, 0, stream>>>(ctx, wtd_v, out, vis_idx, counts + 0, HID, HID, HID);
  gemm_routed<<<gd, 256, 0, stream>>>(ctx, wtd_l, out, lng_idx, counts + 1, HID, HID, HID);
}
